// QuantumDeepField_88562225643966
// MI455X (gfx1250) — matrix operand order repaired and checked by reading, not compiled, not run on hardware
//
#include <hip/hip_runtime.h>

typedef __attribute__((ext_vector_type(16))) _Float16 v16h;
typedef __attribute__((ext_vector_type(8)))  _Float16 v8h;
typedef __attribute__((ext_vector_type(8)))  float    v8f;

#define B_MOL  32
#define G_GRID 2048
#define A_AO   192
#define DIMX   512
#define L_FUNC 3
#define EPSF   1e-12f

// ---------------------------------------------------------------------------
// GTO value: d^(q-1) * exp(-z d^2), q in {1,2,3}
// ---------------------------------------------------------------------------
__device__ __forceinline__ float gto_val(float d, int q, float z) {
    float p = (q <= 1) ? 1.0f : ((q == 2) ? d : d * d);
    return p * __expf(-z * d * d);
}

// Column denominator over the G axis: gden[b*A+a] = max(||GTO col||, eps)
__global__ void k_gto_den(const float* __restrict__ D, const float* __restrict__ zeta,
                          const int* __restrict__ ao, const int* __restrict__ qn,
                          float* __restrict__ gden) {
    int ba = blockIdx.x;                  // b*A + a
    int b  = ba / A_AO;
    int a  = ba - b * A_AO;
    int q  = qn[ba];
    float z = zeta[ao[ba]];
    const float* Dp = D + (size_t)b * G_GRID * A_AO + a;
    float ss = 0.f;
    for (int g = threadIdx.x; g < G_GRID; g += 256) {
        float v = gto_val(Dp[(size_t)g * A_AO], q, z);
        ss += v * v;
    }
    __shared__ float red[256];
    red[threadIdx.x] = ss;
    __syncthreads();
    for (int s = 128; s > 0; s >>= 1) {
        if (threadIdx.x < s) red[threadIdx.x] += red[threadIdx.x + s];
        __syncthreads();
    }
    if (threadIdx.x == 0) gden[ba] = fmaxf(sqrtf(red[0]), EPSF);
}

// Normalized GTOs -> f16  [B,G,A]  (GEMM-1 A matrix, lda = A)
__global__ void k_gto_norm(const float* __restrict__ D, const float* __restrict__ zeta,
                           const int* __restrict__ ao, const int* __restrict__ qn,
                           const float* __restrict__ gden, _Float16* __restrict__ out) {
    int idx = blockIdx.x * 256 + threadIdx.x;          // < B*G*A (exact multiple)
    int a   = idx % A_AO;
    int b   = idx / (G_GRID * A_AO);
    int ba  = b * A_AO + a;
    float v = gto_val(D[idx], qn[ba], zeta[ao[ba]]);
    out[idx] = (_Float16)(v / gden[ba]);
}

// Coefficient column denominator over A: cden[b*DIM+d]
__global__ void k_coef_den(const float* __restrict__ Wc, const int* __restrict__ ao,
                           float* __restrict__ cden) {
    int gid = blockIdx.x * 256 + threadIdx.x;          // b*DIM + d
    int b = gid >> 9;
    int d = gid & (DIMX - 1);
    float ss = 0.f;
    for (int a = 0; a < A_AO; ++a) {
        float w = Wc[(size_t)ao[b * A_AO + a] * DIMX + d];
        ss += w * w;
    }
    cden[gid] = fmaxf(sqrtf(ss), EPSF);
}

// Normalized coefficients -> f16  [B,A,DIM] (GEMM-1 B matrix, ldb = DIM)
__global__ void k_coef_norm(const float* __restrict__ Wc, const int* __restrict__ ao,
                            const float* __restrict__ cden, _Float16* __restrict__ out) {
    int idx = blockIdx.x * 256 + threadIdx.x;          // < B*A*DIM
    int d = idx & (DIMX - 1);
    int t = idx >> 9;
    int a = t % A_AO;
    int b = t / A_AO;
    float w = Wc[(size_t)ao[b * A_AO + a] * DIMX + d];
    out[idx] = (_Float16)(w / cden[b * DIMX + d]);
}

__global__ void k_f32_to_f16(const float* __restrict__ in, _Float16* __restrict__ out, int n) {
    int i = blockIdx.x * 256 + threadIdx.x;
    if (i < n) out[i] = (_Float16)in[i];
}

// ---------------------------------------------------------------------------
// WMMA GEMM: C[M,N] = op(A[M,K] @ B[K,N] (+bias) (relu)) in f16, f32 accum.
// Block: 256 threads = 8 waves. Block tile 128x128, wave tile 32x64.
// Double-buffered LDS software pipeline: global loads of tile k+1 are issued
// before the WMMAs of tile k, staged into the alternate LDS buffer, one
// barrier per k-step. K mult of 32; M mult of 128; N mult of 128.
// ---------------------------------------------------------------------------
__global__ __launch_bounds__(256)
void k_wmma_gemm(const _Float16* __restrict__ Ag_, const _Float16* __restrict__ Bg_,
                 _Float16* __restrict__ Cg_, int K, int lda, int ldb, int ldc,
                 long long sA, long long sB, long long sC,
                 const float* __restrict__ bias, int relu) {
    __shared__ _Float16 Alds[2][128][40];   // [buf][m][k], padded row stride 40
    __shared__ _Float16 Blds[2][128][40];   // [buf][n][k]  (transposed)

    const _Float16* Ag = Ag_ + (size_t)blockIdx.z * sA;
    const _Float16* Bg = Bg_ + (size_t)blockIdx.z * sB;
    _Float16*       Cg = Cg_ + (size_t)blockIdx.z * sC;

    const int tid  = threadIdx.x;
    const int lane = tid & 31;
    const int wave = tid >> 5;
    const int wm   = wave & 3;           // 4 waves along M
    const int wn   = wave >> 2;          // 2 waves along N
    const int lm   = lane & 15;
    const int lhi  = (lane >> 4) & 1;
    const int row0 = blockIdx.x * 128;
    const int col0 = blockIdx.y * 128;

    // per-thread tile-chunk coordinates (fixed for all k-steps)
    // A tile 128x32 = 512 chunks of 8 halves, 2 per thread
    const int ac0r = tid >> 2,           ac0k = (tid & 3) * 8;
    const int ac1r = (tid + 256) >> 2,   ac1k = ((tid + 256) & 3) * 8;
    // B tile 32x128 = 512 chunks of 8 halves (along N), 2 per thread
    const int bc0k = tid >> 4,           bc0n = (tid & 15) * 8;
    const int bc1k = (tid + 256) >> 4,   bc1n = ((tid + 256) & 15) * 8;

    v8f acc[2][4] = {};
    v8h aReg[2], bReg[2];

    auto load_regs = [&](int kt) {
        aReg[0] = *(const v8h*)&Ag[(size_t)(row0 + ac0r) * lda + kt + ac0k];
        aReg[1] = *(const v8h*)&Ag[(size_t)(row0 + ac1r) * lda + kt + ac1k];
        bReg[0] = *(const v8h*)&Bg[(size_t)(kt + bc0k) * ldb + col0 + bc0n];
        bReg[1] = *(const v8h*)&Bg[(size_t)(kt + bc1k) * ldb + col0 + bc1n];
    };
    auto stage_lds = [&](int buf) {
        *(v8h*)&Alds[buf][ac0r][ac0k] = aReg[0];
        *(v8h*)&Alds[buf][ac1r][ac1k] = aReg[1];
        #pragma unroll
        for (int j = 0; j < 8; ++j) Blds[buf][bc0n + j][bc0k] = bReg[0][j];
        #pragma unroll
        for (int j = 0; j < 8; ++j) Blds[buf][bc1n + j][bc1k] = bReg[1][j];
    };

    // pipeline prologue: stage k-tile 0
    load_regs(0);
    stage_lds(0);
    __syncthreads();

    int cur = 0;
    for (int kt = 0; kt < K; kt += 32) {
        const bool has_next = (kt + 32) < K;
        if (has_next) load_regs(kt + 32);               // overlap with WMMAs below
        if ((kt + 64) < K) {                            // GL2 prefetch 2 steps ahead
            __builtin_prefetch(&Ag[(size_t)(row0 + ac0r) * lda + kt + 64 + ac0k], 0, 3);
            __builtin_prefetch(&Bg[(size_t)(kt + 64 + bc0k) * ldb + col0 + bc0n], 0, 3);
        }

        // Fragments in contiguous k order (same for A and B):
        //  A 16x32: lanes 0-15 -> K 0..15; lanes 16-31 -> K 16..31 (row = lane&15)
        //  B 32x16: lanes 0-15 -> K 0..15; lanes 16-31 -> K 16..31 (col = lane&15)
        union Frag { v16h v; v8h h[2]; };
        Frag afrag[2], bfrag[4];
        const int koA = lhi * 16;
        const int koB = lhi * 16;
        #pragma unroll
        for (int mi = 0; mi < 2; ++mi) {
            int r = wm * 32 + mi * 16 + lm;
            afrag[mi].h[0] = *(const v8h*)&Alds[cur][r][koA];
            afrag[mi].h[1] = *(const v8h*)&Alds[cur][r][koA + 8];
        }
        #pragma unroll
        for (int ni = 0; ni < 4; ++ni) {
            int n = wn * 64 + ni * 16 + lm;
            bfrag[ni].h[0] = *(const v8h*)&Blds[cur][n][koB];
            bfrag[ni].h[1] = *(const v8h*)&Blds[cur][n][koB + 8];
        }
        #pragma unroll
        for (int mi = 0; mi < 2; ++mi)
            #pragma unroll
            for (int ni = 0; ni < 4; ++ni)
                acc[mi][ni] = __builtin_amdgcn_wmma_f32_16x16x32_f16(
                    false, afrag[mi].v, false, bfrag[ni].v,
                    (short)0, acc[mi][ni], false, false);

        if (has_next) stage_lds(cur ^ 1);
        __syncthreads();   // next-tile staged; also guards reuse of lds[cur]
        cur ^= 1;
    }

    // Store: C/D layout -> VGPR r: row = r + lhi*8, col = lane&15
    #pragma unroll
    for (int mi = 0; mi < 2; ++mi) {
        #pragma unroll
        for (int ni = 0; ni < 4; ++ni) {
            int rbase = row0 + wm * 32 + mi * 16 + lhi * 8;
            int cc    = col0 + wn * 64 + ni * 16 + lm;
            float bv = bias ? bias[cc] : 0.0f;
            #pragma unroll
            for (int r = 0; r < 8; ++r) {
                float v = acc[mi][ni][r] + bv;
                if (relu) v = fmaxf(v, 0.0f);
                Cg[(size_t)(rbase + r) * ldc + cc] = (_Float16)v;
            }
        }
    }
}

// Per-(b,d) MO scale: sqrt(Ne/DIM) / max(||MO col over G||, eps)
__global__ void k_mo_scalefac(const _Float16* __restrict__ mo, const int* __restrict__ Ne,
                              float* __restrict__ sc) {
    int gid = blockIdx.x * 256 + threadIdx.x;  // b*DIM + d
    int b = gid >> 9;
    const _Float16* p = mo + (size_t)b * G_GRID * DIMX + (gid & (DIMX - 1));
    float ss = 0.f;
    for (int g = 0; g < G_GRID; ++g) {
        float v = (float)p[(size_t)g * DIMX];
        ss += v * v;
    }
    float den = fmaxf(sqrtf(ss), EPSF);
    sc[gid] = sqrtf((float)Ne[b] / (float)DIMX) / den;
}

__global__ void k_mo_scale(_Float16* __restrict__ mo, const float* __restrict__ sc) {
    size_t idx = (size_t)blockIdx.x * 256 + threadIdx.x;   // < B*G*DIM
    int d = (int)(idx & (DIMX - 1));
    int b = (int)(idx >> 20);                              // G*DIM = 2^20
    mo[idx] = (_Float16)((float)mo[idx] * sc[b * DIMX + d]);
}

// Sum-pool over G fused with final [DIM]->1 projection
__global__ void k_pool(const _Float16* __restrict__ v, const float* __restrict__ Wp,
                       const float* __restrict__ Wpb, float* __restrict__ E) {
    int b = blockIdx.x, tid = threadIdx.x;
    const _Float16* p = v + (size_t)b * G_GRID * DIMX;
    float p0 = 0.f, p1 = 0.f;
    for (int g = 0; g < G_GRID; ++g) {
        p0 += (float)p[(size_t)g * DIMX + tid];
        p1 += (float)p[(size_t)g * DIMX + tid + 256];
    }
    float partial = p0 * Wp[tid] + p1 * Wp[tid + 256];
    __shared__ float red[256];
    red[tid] = partial;
    __syncthreads();
    for (int s = 128; s > 0; s >>= 1) {
        if (tid < s) red[tid] += red[tid + s];
        __syncthreads();
    }
    if (tid == 0) E[b] = red[0] + Wpb[0];
}

// ---------------------------------------------------------------------------
extern "C" void kernel_launch(void* const* d_in, const int* in_sizes, int n_in,
                              void* d_out, int out_size, void* d_ws, size_t ws_size,
                              hipStream_t stream) {
    (void)in_sizes; (void)n_in; (void)out_size; (void)ws_size;
    const float* D    = (const float*)d_in[0];   // [B,G,A]
    const float* Wc   = (const float*)d_in[1];   // [400,512]
    const float* Zeta = (const float*)d_in[2];   // [400,1]
    const float* WfW  = (const float*)d_in[3];   // [3,512,512]
    const float* WfB  = (const float*)d_in[4];   // [3,512]
    const float* WpW  = (const float*)d_in[5];   // [512,1]
    const float* WpB  = (const float*)d_in[6];   // [1]
    const int*   AO   = (const int*)d_in[7];     // [B,A]
    const int*   QN   = (const int*)d_in[8];     // [B,A]
    const int*   NE   = (const int*)d_in[9];     // [B]
    float* Out = (float*)d_out;                  // [B,1]

    // ---- workspace carving (256-byte aligned) ----
    char* ws = (char*)d_ws;
    size_t off = 0;
    auto carve = [&](size_t bytes) -> char* {
        off = (off + 255) & ~(size_t)255;
        char* p = ws + off;
        off += bytes;
        return p;
    };
    float*    gden   = (float*)   carve((size_t)B_MOL * A_AO * 4);
    float*    cden   = (float*)   carve((size_t)B_MOL * DIMX * 4);
    float*    mosc   = (float*)   carve((size_t)B_MOL * DIMX * 4);
    _Float16* gto16  = (_Float16*)carve((size_t)B_MOL * G_GRID * A_AO * 2);
    _Float16* coef16 = (_Float16*)carve((size_t)B_MOL * A_AO * DIMX * 2);
    _Float16* w16    = (_Float16*)carve((size_t)L_FUNC * DIMX * DIMX * 2);
    _Float16* bufX   = (_Float16*)carve((size_t)B_MOL * G_GRID * DIMX * 2);
    _Float16* bufY   = (_Float16*)carve((size_t)B_MOL * G_GRID * DIMX * 2);

    const long long BG = (long long)B_MOL * G_GRID;

    // 1) GTO denominators + normalized GTOs (f16)
    k_gto_den<<<B_MOL * A_AO, 256, 0, stream>>>(D, Zeta, AO, QN, gden);
    k_gto_norm<<<(B_MOL * G_GRID * A_AO) / 256, 256, 0, stream>>>(D, Zeta, AO, QN, gden, gto16);

    // 2) Coefficient denominators + normalized coefs (f16)
    k_coef_den<<<(B_MOL * DIMX) / 256, 256, 0, stream>>>(Wc, AO, cden);
    k_coef_norm<<<(B_MOL * A_AO * DIMX) / 256, 256, 0, stream>>>(Wc, AO, cden, coef16);

    // 3) MLP weights -> f16
    k_f32_to_f16<<<(L_FUNC * DIMX * DIMX) / 256, 256, 0, stream>>>(WfW, w16, L_FUNC * DIMX * DIMX);

    // 4) MOs = n_GTOs @ coefs  (batched: 32 x [2048x192x512])
    k_wmma_gemm<<<dim3(G_GRID / 128, DIMX / 128, B_MOL), 256, 0, stream>>>(
        gto16, coef16, bufX, /*K=*/A_AO, /*lda=*/A_AO, /*ldb=*/DIMX, /*ldc=*/DIMX,
        (long long)G_GRID * A_AO, (long long)A_AO * DIMX, (long long)G_GRID * DIMX,
        nullptr, 0);

    // 5) MO normalization & electron-count scaling
    k_mo_scalefac<<<(B_MOL * DIMX) / 256, 256, 0, stream>>>(bufX, NE, mosc);
    k_mo_scale<<<(unsigned)((BG * DIMX) / 256), 256, 0, stream>>>(bufX, mosc);

    // 6) 3-layer MLP: [B*G,512] @ [512,512] + bias, ReLU  (shared weights -> flat M)
    dim3 gdnn((unsigned)(BG / 128), DIMX / 128, 1);
    k_wmma_gemm<<<gdnn, 256, 0, stream>>>(bufX, w16 + 0 * DIMX * DIMX, bufY,
        DIMX, DIMX, DIMX, DIMX, 0, 0, 0, WfB + 0 * DIMX, 1);
    k_wmma_gemm<<<gdnn, 256, 0, stream>>>(bufY, w16 + 1 * DIMX * DIMX, bufX,
        DIMX, DIMX, DIMX, DIMX, 0, 0, 0, WfB + 1 * DIMX, 1);
    k_wmma_gemm<<<gdnn, 256, 0, stream>>>(bufX, w16 + 2 * DIMX * DIMX, bufY,
        DIMX, DIMX, DIMX, DIMX, 0, 0, 0, WfB + 2 * DIMX, 1);

    // 7) sum-pool over G + final projection
    k_pool<<<B_MOL, 256, 0, stream>>>(bufY, WpW, WpB, Out);
}
